// SAGE_edge_aff_4191888081216
// MI455X (gfx1250) — matrix operand order repaired and checked by reading, not compiled, not run on hardware
//
#include <hip/hip_runtime.h>
#include <hip/hip_bf16.h>

typedef __attribute__((ext_vector_type(16))) _Float16 v16h;
typedef __attribute__((ext_vector_type(8)))  _Float16 v8h;
typedef __attribute__((ext_vector_type(8)))  float    v8f;

#define SLOPE 0.01f

// ---------------------------------------------------------------------------
// A-fragment loader: contiguous-k 16-bit 16x32 layout (v16h per lane).
// lanes 0-15 hold K[0..16), lanes 16-31 hold K[16..32).
// ---------------------------------------------------------------------------
__device__ __forceinline__ v16h load_afrag(const _Float16* __restrict__ row,
                                           int kb, int half) {
    v8h lo = *reinterpret_cast<const v8h*>(row + kb + 16 * half);
    v8h hi = *reinterpret_cast<const v8h*>(row + kb + 16 * half + 8);
    v16h a;
#pragma unroll
    for (int i = 0; i < 8; ++i) { a[i] = lo[i]; a[8 + i] = hi[i]; }
    return a;
}

// ---------------------------------------------------------------------------
// WMMA GEMM (f16 inputs, f32 accumulate):
//   out[M,O] = act( A1[M,K] @ W1[O,K]^T  (+ A2[M,K] @ W2[O,K]^T if DUAL) + bias )
// Also writes an f16 mirror (outh) for the next GEMM stage when non-null.
// Wave tile: 16 rows x 64 cols (4 independent accumulators, A reused 4x).
// ACT: 0 = none, 1 = relu, 2 = leaky(0.01)
// ---------------------------------------------------------------------------
template <int ACT, bool DUAL>
__global__ void gemm_wmma_kernel(const _Float16* __restrict__ A1, const _Float16* __restrict__ W1,
                                 const _Float16* __restrict__ A2, const _Float16* __restrict__ W2,
                                 const float* __restrict__ bias,
                                 float* __restrict__ out, _Float16* __restrict__ outh,
                                 int M, int K, int O) {
    const int lane = threadIdx.x & 31;
    const int wave = blockIdx.x * (blockDim.x >> 5) + (threadIdx.x >> 5);
    const int mTiles  = (M + 15) >> 4;
    const int nBlocks = O >> 6;                     // 64-column blocks
    if (wave >= mTiles * nBlocks) return;
    const int mt = wave % mTiles;
    const int nb = wave / mTiles;
    const int m0 = mt << 4;
    const int n0 = nb << 6;
    const int half = lane >> 4;                     // wave32 half
    const int l15  = lane & 15;

    int mA = m0 + l15;                              // A row for this lane
    if (mA >= M) mA = M - 1;                        // clamp (stores are guarded)
    const _Float16* a1row = A1 + (size_t)mA * K;
    const _Float16* a2row = DUAL ? (A2 + (size_t)mA * K) : nullptr;

    v8f acc[4];
#pragma unroll
    for (int i = 0; i < 4; ++i) acc[i] = (v8f){0.f,0.f,0.f,0.f,0.f,0.f,0.f,0.f};

    for (int kb = 0; kb < K; kb += 32) {
        v16h a1 = load_afrag(a1row, kb, half);
        v16h a2v;
        if (DUAL) a2v = load_afrag(a2row, kb, half);
#pragma unroll
        for (int cb = 0; cb < 4; ++cb) {
            const int n = n0 + cb * 16 + l15;       // B column (row of W)
            // B fragment: 16 contiguous K values starting at 16*half (2x b128)
            v16h b = *reinterpret_cast<const v16h*>(W1 + (size_t)n * K + kb + 16 * half);
            acc[cb] = __builtin_amdgcn_wmma_f32_16x16x32_f16(
                false, a1, false, b, (short)0, acc[cb], false, false);
            if (DUAL) {
                v16h b2 = *reinterpret_cast<const v16h*>(W2 + (size_t)n * K + kb + 16 * half);
                acc[cb] = __builtin_amdgcn_wmma_f32_16x16x32_f16(
                    false, a2v, false, b2, (short)0, acc[cb], false, false);
            }
        }
    }

    // --- epilogue: bias + activation + guarded stores (C/D layout) ---
#pragma unroll
    for (int cb = 0; cb < 4; ++cb) {
        const int n = n0 + cb * 16 + l15;
        const float bv = bias ? bias[n] : 0.0f;
#pragma unroll
        for (int r = 0; r < 8; ++r) {
            const int m = m0 + r + 8 * half;
            if (m < M) {
                float v = acc[cb][r] + bv;
                if (ACT == 1)      v = v > 0.f ? v : 0.f;
                else if (ACT == 2) v = v > 0.f ? v : SLOPE * v;
                out[(size_t)m * O + n] = v;
                if (outh) outh[(size_t)m * O + n] = (_Float16)v;
            }
        }
    }
}

// ---------------------------------------------------------------------------
// Helper kernels
// ---------------------------------------------------------------------------
__global__ void cvt_f16_kernel(const float* __restrict__ src, _Float16* __restrict__ dst, int n) {
    int i = blockIdx.x * blockDim.x + threadIdx.x;
    if (i < n) dst[i] = (_Float16)src[i];
}

__global__ void zero_kernel(float* p, int n) {
    int i = blockIdx.x * blockDim.x + threadIdx.x;
    if (i < n) p[i] = 0.0f;
}

__global__ void deg_kernel(const int* __restrict__ ei, float* __restrict__ deg, int E) {
    int e = blockIdx.x * blockDim.x + threadIdx.x;
    if (e < E) atomicAdd(&deg[ei[E + e]], 1.0f);     // dst = ei[1][e]
}

// agg[dst] += h[src], 4 features per thread
__global__ void scatter_kernel(const float* __restrict__ h, const int* __restrict__ ei,
                               float* __restrict__ agg, int E, int H) {
    const int chunks = H >> 2;
    int t = blockIdx.x * blockDim.x + threadIdx.x;
    if (t >= E * chunks) return;
    int e = t / chunks;
    int c = (t - e * chunks) << 2;
    int s = ei[e];                                   // src = ei[0][e]
    int d = ei[E + e];                               // dst = ei[1][e]
    const float* hp = h + (size_t)s * H + c;
    float* ap = agg + (size_t)d * H + c;
    atomicAdd(ap + 0, hp[0]);
    atomicAdd(ap + 1, hp[1]);
    atomicAdd(ap + 2, hp[2]);
    atomicAdd(ap + 3, hp[3]);
}

// mean in f16: aggh[i] = (agg[i] / max(deg,1)) — consumed only by the WMMA GEMM
__global__ void mean_kernel(const float* __restrict__ agg, const float* __restrict__ deg,
                            _Float16* __restrict__ aggh, int total, int H) {
    int i = blockIdx.x * blockDim.x + threadIdx.x;
    if (i < total) {
        float d = deg[i / H];
        aggh[i] = (_Float16)(agg[i] / fmaxf(d, 1.0f));
    }
}

// Per-node rank-1 projections of the edge MLP (one wave32 per node, h = [N,128])
__global__ void node_scalars_kernel(const float* __restrict__ h,
                                    const float* __restrict__ w96,  // [192]
                                    const float* __restrict__ w32,  // [64]
                                    float* __restrict__ s96l, float* __restrict__ s96r,
                                    float* __restrict__ s32l, float* __restrict__ s32r,
                                    int n) {
    int node = blockIdx.x * (blockDim.x >> 5) + (threadIdx.x >> 5);
    if (node >= n) return;
    int lane = threadIdx.x & 31;
    const float* hp = h + (size_t)node * 128;
    float h0 = hp[lane], h1 = hp[lane + 32], h2 = hp[lane + 64], h3 = hp[lane + 96];
    float p96l = h0 * w96[lane] + h1 * w96[lane + 32] + h2 * w96[lane + 64];
    float p96r = h0 * w96[96 + lane] + h1 * w96[128 + lane] + h2 * w96[160 + lane];
    float p32l = h3 * w32[lane];
    float p32r = h3 * w32[32 + lane];
#pragma unroll
    for (int off = 16; off > 0; off >>= 1) {
        p96l += __shfl_xor(p96l, off, 32);
        p96r += __shfl_xor(p96r, off, 32);
        p32l += __shfl_xor(p32l, off, 32);
        p32r += __shfl_xor(p32r, off, 32);
    }
    if (lane == 0) {
        s96l[node] = p96l; s96r[node] = p96r;
        s32l[node] = p32l; s32r[node] = p32r;
    }
}

__global__ void edge_vals_kernel(const int* __restrict__ ei, const float* __restrict__ attr,
                                 const float* __restrict__ s96l, const float* __restrict__ s96r,
                                 const float* __restrict__ s32l, const float* __restrict__ s32r,
                                 const float* __restrict__ b96p, const float* __restrict__ b32p,
                                 float* __restrict__ ev, int E) {
    int e = blockIdx.x * blockDim.x + threadIdx.x;
    if (e >= E) return;
    int s = ei[e], d = ei[E + e];
    float f96 = s96l[s] + s96r[d] + b96p[0];
    float f32 = s32l[s] + s32r[d] + b32p[0];
    f96 = f96 > 0.f ? f96 : 0.f;
    f32 = f32 > 0.f ? f32 : 0.f;
    ev[e] = f96 * attr[e] + f32;
}

__global__ void group_mean_kernel(const float* __restrict__ ev, float* __restrict__ out, int G) {
    int g = blockIdx.x * blockDim.x + threadIdx.x;
    if (g >= G) return;
    float s = 0.f;
    const float* p = ev + (size_t)g * 48;
#pragma unroll
    for (int i = 0; i < 48; ++i) s += p[i];
    out[g] = s * (1.0f / 48.0f);
}

// ---------------------------------------------------------------------------
// Host-side launch plumbing
// ---------------------------------------------------------------------------
static inline void gemm(int act, bool dual,
                        const _Float16* A1, const _Float16* W1,
                        const _Float16* A2, const _Float16* W2,
                        const float* bias, float* out, _Float16* outh,
                        int M, int K, int O, hipStream_t s) {
    int mTiles = (M + 15) >> 4;
    int waves = mTiles * (O >> 6);
    dim3 grid((waves + 7) / 8), block(256);
    if (dual) {
        if (act == 1)      gemm_wmma_kernel<1, true><<<grid, block, 0, s>>>(A1, W1, A2, W2, bias, out, outh, M, K, O);
        else if (act == 2) gemm_wmma_kernel<2, true><<<grid, block, 0, s>>>(A1, W1, A2, W2, bias, out, outh, M, K, O);
        else               gemm_wmma_kernel<0, true><<<grid, block, 0, s>>>(A1, W1, A2, W2, bias, out, outh, M, K, O);
    } else {
        if (act == 1)      gemm_wmma_kernel<1, false><<<grid, block, 0, s>>>(A1, W1, nullptr, nullptr, bias, out, outh, M, K, O);
        else if (act == 2) gemm_wmma_kernel<2, false><<<grid, block, 0, s>>>(A1, W1, nullptr, nullptr, bias, out, outh, M, K, O);
        else               gemm_wmma_kernel<0, false><<<grid, block, 0, s>>>(A1, W1, nullptr, nullptr, bias, out, outh, M, K, O);
    }
}

static inline void cvt(const float* src, _Float16* dst, int n, hipStream_t s) {
    cvt_f16_kernel<<<(n + 255) / 256, 256, 0, s>>>(src, dst, n);
}

// scatter-mean: aggh(f16) = mean over incoming edges of h[src] (h in f32)
static inline void aggregate(const float* h, const int* ei,
                             float* agg, float* deg, _Float16* aggh,
                             int Nn, int E, int H, hipStream_t s) {
    zero_kernel<<<(Nn * H + 255) / 256, 256, 0, s>>>(agg, Nn * H);
    zero_kernel<<<(Nn + 255) / 256, 256, 0, s>>>(deg, Nn);
    deg_kernel<<<(E + 255) / 256, 256, 0, s>>>(ei, deg, E);
    int scat = E * (H >> 2);
    scatter_kernel<<<(scat + 255) / 256, 256, 0, s>>>(h, ei, agg, E, H);
    mean_kernel<<<(Nn * H + 255) / 256, 256, 0, s>>>(agg, deg, aggh, Nn * H, H);
}

extern "C" void kernel_launch(void* const* d_in, const int* in_sizes, int n_in,
                              void* d_out, int out_size, void* d_ws, size_t ws_size,
                              hipStream_t stream) {
    (void)n_in; (void)out_size; (void)ws_size;

    const float* x     = (const float*)d_in[0];
    const int*   ei    = (const int*)d_in[1];
    const float* attr  = (const float*)d_in[2];
    const float* pre_w = (const float*)d_in[4];
    const float* pre_b = (const float*)d_in[5];
    const float* c1_lw = (const float*)d_in[6];
    const float* c1_lb = (const float*)d_in[7];
    const float* c1_rw = (const float*)d_in[8];
    const float* c2_lw = (const float*)d_in[9];
    const float* c2_lb = (const float*)d_in[10];
    const float* c2_rw = (const float*)d_in[11];
    const float* c3_lw = (const float*)d_in[12];
    const float* c3_lb = (const float*)d_in[13];
    const float* c3_rw = (const float*)d_in[14];
    const float* hh1_w = (const float*)d_in[15];
    const float* hh1_b = (const float*)d_in[16];
    const float* hh2_w = (const float*)d_in[17];
    const float* hh2_b = (const float*)d_in[18];
    const float* oo_w  = (const float*)d_in[19];
    const float* oo_b  = (const float*)d_in[20];
    const float* w96   = (const float*)d_in[21];
    const float* b96   = (const float*)d_in[22];
    const float* w32   = (const float*)d_in[23];
    const float* b32   = (const float*)d_in[24];

    const int Nn = in_sizes[0] / 128;   // 25000 nodes
    const int E  = in_sizes[1] / 2;     // 600000 edges
    const int G  = E / 48;
    const int H  = 256;

    // ---- workspace layout ----
    char* base = (char*)d_ws;
    size_t off = 0;
    auto alloc = [&](size_t bytes) {
        void* p = base + off;
        off += (bytes + 255) & ~(size_t)255;
        return p;
    };
    // f32 buffers
    float* P0   = (float*)alloc((size_t)Nn * H * sizeof(float));
    float* P1   = (float*)alloc((size_t)Nn * H * sizeof(float));   // agg sums
    float* P2   = (float*)alloc((size_t)Nn * H * sizeof(float));
    float* deg  = (float*)alloc((size_t)Nn * sizeof(float));
    float* s96l = (float*)alloc((size_t)Nn * sizeof(float));
    float* s96r = (float*)alloc((size_t)Nn * sizeof(float));
    float* s32l = (float*)alloc((size_t)Nn * sizeof(float));
    float* s32r = (float*)alloc((size_t)Nn * sizeof(float));
    float* ev   = (float*)alloc((size_t)E * sizeof(float));
    // f16 mirrors
    _Float16* Xh   = (_Float16*)alloc((size_t)Nn * 128 * sizeof(_Float16));
    _Float16* P0h  = (_Float16*)alloc((size_t)Nn * H * sizeof(_Float16));
    _Float16* P2h  = (_Float16*)alloc((size_t)Nn * H * sizeof(_Float16));
    _Float16* AGGh = (_Float16*)alloc((size_t)Nn * H * sizeof(_Float16));
    // f16 weights
    _Float16* preWh  = (_Float16*)alloc(256 * 128 * sizeof(_Float16));
    _Float16* c1lwH  = (_Float16*)alloc(256 * 256 * sizeof(_Float16));
    _Float16* c1rwH  = (_Float16*)alloc(256 * 256 * sizeof(_Float16));
    _Float16* c2lwH  = (_Float16*)alloc(256 * 256 * sizeof(_Float16));
    _Float16* c2rwH  = (_Float16*)alloc(256 * 256 * sizeof(_Float16));
    _Float16* c3lwH  = (_Float16*)alloc(128 * 256 * sizeof(_Float16));
    _Float16* c3rwH  = (_Float16*)alloc(128 * 256 * sizeof(_Float16));
    _Float16* hh1Wh  = (_Float16*)alloc(256 * 256 * sizeof(_Float16));
    _Float16* hh2Wh  = (_Float16*)alloc(256 * 256 * sizeof(_Float16));
    _Float16* ooWh   = (_Float16*)alloc(128 * 128 * sizeof(_Float16));

    // ---- 0. one-time f16 staging of inputs/weights ----
    cvt(x, Xh, Nn * 128, stream);
    cvt(pre_w, preWh, 256 * 128, stream);
    cvt(c1_lw, c1lwH, 256 * 256, stream);
    cvt(c1_rw, c1rwH, 256 * 256, stream);
    cvt(c2_lw, c2lwH, 256 * 256, stream);
    cvt(c2_rw, c2rwH, 256 * 256, stream);
    cvt(c3_lw, c3lwH, 128 * 256, stream);
    cvt(c3_rw, c3rwH, 128 * 256, stream);
    cvt(hh1_w, hh1Wh, 256 * 256, stream);
    cvt(hh2_w, hh2Wh, 256 * 256, stream);
    cvt(oo_w, ooWh, 128 * 128, stream);

    // ---- 1. prelin: P0 = relu(x @ pre_w^T + pre_b)   [N,128] -> [N,256] ----
    gemm(1, false, Xh, preWh, nullptr, nullptr, pre_b, P0, P0h, Nn, 128, 256, stream);

    // ---- 2. conv1 -> P2 ; hh1 -> P0 ----
    aggregate(P0, ei, P1, deg, AGGh, Nn, E, 256, stream);
    gemm(1, true, AGGh, c1lwH, P0h, c1rwH, c1_lb, P2, P2h, Nn, 256, 256, stream);
    gemm(2, false, P2h, hh1Wh, nullptr, nullptr, hh1_b, P0, P0h, Nn, 256, 256, stream);

    // ---- 3. conv2 -> P2 ; hh2 -> P0 ----
    aggregate(P0, ei, P1, deg, AGGh, Nn, E, 256, stream);
    gemm(1, true, AGGh, c2lwH, P0h, c2rwH, c2_lb, P2, P2h, Nn, 256, 256, stream);
    gemm(2, false, P2h, hh2Wh, nullptr, nullptr, hh2_b, P0, P0h, Nn, 256, 256, stream);

    // ---- 4. conv3 -> P2 [N,128] ; oo -> P0 [N,128] ----
    aggregate(P0, ei, P1, deg, AGGh, Nn, E, 256, stream);
    gemm(1, true, AGGh, c3lwH, P0h, c3rwH, c3_lb, P2, P2h, Nn, 256, 128, stream);
    gemm(2, false, P2h, ooWh, nullptr, nullptr, oo_b, P0, nullptr, Nn, 128, 128, stream);

    // ---- 5. edge MLP (rank-1 factorized) + group-of-48 mean ----
    node_scalars_kernel<<<(Nn + 7) / 8, 256, 0, stream>>>(P0, w96, w32,
                                                          s96l, s96r, s32l, s32r, Nn);
    edge_vals_kernel<<<(E + 255) / 256, 256, 0, stream>>>(ei, attr, s96l, s96r, s32l, s32r,
                                                          b96, b32, ev, E);
    group_mean_kernel<<<(G + 255) / 256, 256, 0, stream>>>(ev, (float*)d_out, G);
}
